// TemporalLobe_26757646254135
// MI455X (gfx1250) — matrix operand order repaired and checked by reading, not compiled, not run on hardware
//
#include <hip/hip_runtime.h>

// Problem constants (from reference): B=64, D=1024, H=1024, V=32000, T=20
constexpr int kB = 64;
constexpr int kD = 1024;
constexpr int kH = 1024;
constexpr int kV = 32000;
constexpr int kT = 20;

typedef __attribute__((ext_vector_type(16))) __bf16 v16bf;
typedef __attribute__((ext_vector_type(8)))  __bf16 v8bf;
typedef __attribute__((ext_vector_type(8)))  float  v8f;

__device__ __forceinline__ unsigned short f32_to_bf16_rne(float f) {
  unsigned u = __float_as_uint(f);
  u += 0x7FFFu + ((u >> 16) & 1u);   // round-to-nearest-even
  return (unsigned short)(u >> 16);
}

// ---------------------------------------------------------------------------
// One-time per call: W_out f32 -> bf16 (halves the dominant per-step stream;
// 65.5MB result stays resident in the 192MB L2 across all 20 steps).
// ---------------------------------------------------------------------------
__global__ void k_cvt_bf16(const float* __restrict__ src,
                           unsigned short* __restrict__ dst, int n) {
  int i = blockIdx.x * blockDim.x + threadIdx.x;
  if (i < n) dst[i] = f32_to_bf16_rne(src[i]);
}

// ---------------------------------------------------------------------------
// h = latent @ W_proj.T + b_proj   (exact f32; bandwidth-bound on 4MB W_proj)
// block = 64 lanes (one per batch row), each block computes 8 j-columns.
// ---------------------------------------------------------------------------
__global__ void k_init_hidden(const float* __restrict__ latent,
                              const float* __restrict__ Wp,
                              const float* __restrict__ bp,
                              float* __restrict__ h) {
  const int b  = threadIdx.x;        // 0..63
  const int j0 = blockIdx.x * 8;     // H/8 = 128 blocks
  float acc[8];
#pragma unroll
  for (int jj = 0; jj < 8; ++jj) acc[jj] = bp[j0 + jj];
  const float* lv = latent + b * kD;
  for (int k = 0; k < kD; ++k) {
    float x = lv[k];
#pragma unroll
    for (int jj = 0; jj < 8; ++jj)
      acc[jj] += Wp[(size_t)(j0 + jj) * kD + k] * x;   // uniform addr -> SMEM
  }
#pragma unroll
  for (int jj = 0; jj < 8; ++jj) h[b * kH + j0 + jj] = acc[jj];
}

__global__ void k_init_tok(int* __restrict__ tok) {
  if (threadIdx.x < kB) tok[threadIdx.x] = 0;   // <SOS> = 0
}

// ---------------------------------------------------------------------------
// Fused GRU step: x = emb[tok]; gi = x@W_ih.T+b_ih; gh = h@W_hh.T+b_hh;
// gates + h update, all in exact f32 (20-step recurrence -> keep precision).
// Also emits bf16 copy of h for the WMMA logits GEMM.
// block = 64 lanes (batch), 8 j-columns per block => 128 blocks/step.
// Weight traffic = full W_ih+W_hh (25MB) once per step; x/h reads are
// L2/WGP$-resident (512KB unique).
// ---------------------------------------------------------------------------
__global__ void k_gru(const float* __restrict__ emb, const int* __restrict__ tok,
                      const float* __restrict__ Wih, const float* __restrict__ bih,
                      const float* __restrict__ Whh, const float* __restrict__ bhh,
                      const float* __restrict__ hin, float* __restrict__ hout,
                      unsigned short* __restrict__ hbf) {
  const int b  = threadIdx.x;
  const int j0 = blockIdx.x * 8;
  const float* x  = emb + (size_t)tok[b] * kH;
  const float* hv = hin + b * kH;
  float ir[8], iz[8], inn[8], hr[8], hz[8], hn[8];
#pragma unroll
  for (int jj = 0; jj < 8; ++jj) {
    ir[jj]  = bih[j0 + jj];
    iz[jj]  = bih[kH + j0 + jj];
    inn[jj] = bih[2 * kH + j0 + jj];
    hr[jj]  = bhh[j0 + jj];
    hz[jj]  = bhh[kH + j0 + jj];
    hn[jj]  = bhh[2 * kH + j0 + jj];
  }
  const size_t plane = (size_t)kH * kH;
  for (int k = 0; k < kH; ++k) {
    float xv  = x[k];
    float hvk = hv[k];
#pragma unroll
    for (int jj = 0; jj < 8; ++jj) {
      size_t r0 = (size_t)(j0 + jj) * kH + k;
      ir[jj]  += Wih[r0] * xv;
      iz[jj]  += Wih[plane + r0] * xv;
      inn[jj] += Wih[2 * plane + r0] * xv;
      hr[jj]  += Whh[r0] * hvk;
      hz[jj]  += Whh[plane + r0] * hvk;
      hn[jj]  += Whh[2 * plane + r0] * hvk;
    }
  }
#pragma unroll
  for (int jj = 0; jj < 8; ++jj) {
    float r = 1.f / (1.f + expf(-(ir[jj] + hr[jj])));
    float z = 1.f / (1.f + expf(-(iz[jj] + hz[jj])));
    float n = tanhf(inn[jj] + r * hn[jj]);
    float hnew = (1.f - z) * n + z * hv[j0 + jj];
    hout[b * kH + j0 + jj] = hnew;
    hbf[b * kH + j0 + jj]  = f32_to_bf16_rne(hnew);
  }
}

// ---------------------------------------------------------------------------
// logits = h_bf16 @ W_out_bf16.T + b_out via v_wmma_f32_16x16x32_bf16.
// M-blocked: one wave computes the full 64(M) x 16(N) strip for its N-tile:
//   - B fragment (one contiguous 32B run of a W_out row per lane) is loaded
//     ONCE per K-iter and reused across 4 WMMAs -> B L2 traffic = 65.5MB/step
//     (vs 262MB with per-tile waves). B cannot fit in WGP$, so this is the
//     stream that sets the ceiling.
//   - A (h bf16, 128KB total) is re-read per block but is WGP$/L2-resident.
// A layout (contiguous K, same order as B): lanes 0-15 row M=l15 get K 0..15,
// lanes 16-31 get K 16..31  -> two 16B loads per fragment.
// B layout (K x N): lane holds column n=l15, K base = half*16, packed 2/VGPR
// -> one contiguous 32B load per lane.
// ---------------------------------------------------------------------------
__global__ void k_logits_wmma(const unsigned short* __restrict__ hbf_u,
                              const unsigned short* __restrict__ wbf_u,
                              const float* __restrict__ bout,
                              float* __restrict__ logits) {
  const __bf16* hb = reinterpret_cast<const __bf16*>(hbf_u);
  const __bf16* wb = reinterpret_cast<const __bf16*>(wbf_u);
  const int lane  = threadIdx.x;        // 0..31 (one wave per block)
  const int ntile = blockIdx.x;         // 0..V/16-1
  const int half  = lane >> 4;          // 0 or 1
  const int l15   = lane & 15;

  const __bf16* Brow  = wb + (size_t)(ntile * 16 + l15) * kH + half * 16;
  const __bf16* Arow0 = hb + (size_t)l15 * kH + half * 16;   // M-tile stride 16*kH

  v8f c0 = {}, c1 = {}, c2 = {}, c3 = {};
  for (int k0 = 0; k0 < kH; k0 += 32) {
    v16bf Bv = *(const v16bf*)(Brow + k0);          // 16 contiguous bf16 (32B)
    __builtin_prefetch(Brow + k0 + 256, 0, 1);      // global_prefetch_b8

    const __bf16* a0 = Arow0 + k0;
#pragma unroll
    for (int mt = 0; mt < 4; ++mt) {
      const __bf16* ab = a0 + (size_t)mt * 16 * kH;
      v8bf  alo = *(const v8bf*)(ab);               // K base..base+7
      v8bf  ahi = *(const v8bf*)(ab + 8);           // K base+8..base+15
      v16bf A   = __builtin_shufflevector(alo, ahi,
                    0, 1, 2, 3, 4, 5, 6, 7, 8, 9, 10, 11, 12, 13, 14, 15);
      v8f& c = (mt == 0) ? c0 : (mt == 1) ? c1 : (mt == 2) ? c2 : c3;
      c = __builtin_amdgcn_wmma_f32_16x16x32_bf16(
          /*neg_a=*/false, A, /*neg_b=*/false, Bv,
          /*c_mod=*/(short)0, c, /*reuse_a=*/false, /*reuse_b=*/false);
    }
  }

  // C/D layout: lane = N column (l15), VGPR r -> M = mt*16 + r + half*8
  const int n    = ntile * 16 + l15;
  const float bb = bout[n];
#pragma unroll
  for (int mt = 0; mt < 4; ++mt) {
    const v8f& c = (mt == 0) ? c0 : (mt == 1) ? c1 : (mt == 2) ? c2 : c3;
#pragma unroll
    for (int r = 0; r < 8; ++r) {
      int m = mt * 16 + half * 8 + r;
      logits[(size_t)m * kV + n] = c[r] + bb;
    }
  }
}

// ---------------------------------------------------------------------------
// Per-batch-row argmax over V=32000, first-occurrence tie-break (jnp.argmax).
// Writes next token (int, workspace) and the float token into d_out[b*T + t].
// ---------------------------------------------------------------------------
__global__ void k_argmax(const float* __restrict__ logits, int* __restrict__ tok,
                         float* __restrict__ out, int t) {
  __shared__ float sv[256];
  __shared__ int   si[256];
  const int b   = blockIdx.x;
  const int tid = threadIdx.x;
  const float* row = logits + (size_t)b * kV;
  float best = -3.4e38f;
  int   bi   = 0;
  for (int i = tid; i < kV; i += 256) {
    float v = row[i];
    if (v > best) { best = v; bi = i; }   // strictly greater -> keeps first
  }
  sv[tid] = best; si[tid] = bi;
  __syncthreads();
  for (int s = 128; s > 0; s >>= 1) {
    if (tid < s) {
      float v2 = sv[tid + s]; int i2 = si[tid + s];
      if (v2 > sv[tid] || (v2 == sv[tid] && i2 < si[tid])) {
        sv[tid] = v2; si[tid] = i2;
      }
    }
    __syncthreads();
  }
  if (tid == 0) {
    tok[b] = si[0];
    out[b * kT + t] = (float)si[0];
  }
}

// Final hidden -> d_out tail (reference returns h_final[1,B,H] after tokens)
__global__ void k_copy_h(const float* __restrict__ h, float* __restrict__ out) {
  int i = blockIdx.x * blockDim.x + threadIdx.x;   // B*H threads
  out[kB * kT + i] = h[i];
}

// ---------------------------------------------------------------------------
extern "C" void kernel_launch(void* const* d_in, const int* in_sizes, int n_in,
                              void* d_out, int out_size, void* d_ws, size_t ws_size,
                              hipStream_t stream) {
  (void)in_sizes; (void)n_in; (void)out_size; (void)ws_size;
  const float* latent = (const float*)d_in[0];
  const float* Wproj  = (const float*)d_in[1];
  const float* bproj  = (const float*)d_in[2];
  const float* emb    = (const float*)d_in[3];
  const float* Wih    = (const float*)d_in[4];
  const float* bih    = (const float*)d_in[5];
  const float* Whh    = (const float*)d_in[6];
  const float* bhh    = (const float*)d_in[7];
  const float* Wout   = (const float*)d_in[8];
  const float* bout   = (const float*)d_in[9];
  float* out = (float*)d_out;

  // Workspace layout (~74.4MB): all offsets 256B-aligned.
  char* ws = (char*)d_ws;
  size_t off = 0;
  unsigned short* wbf = (unsigned short*)(ws + off); off += (size_t)kV * kH * 2;     // 65,536,000
  float* logits       = (float*)(ws + off);          off += (size_t)kB * kV * 4;     //  8,192,000
  float* hA           = (float*)(ws + off);          off += (size_t)kB * kH * 4;     //    262,144
  float* hB           = (float*)(ws + off);          off += (size_t)kB * kH * 4;     //    262,144
  unsigned short* hbf = (unsigned short*)(ws + off); off += (size_t)kB * kH * 2;     //    131,072
  int* tok            = (int*)(ws + off);

  // One-time per call
  k_cvt_bf16<<<(kV * kH + 255) / 256, 256, 0, stream>>>(Wout, wbf, kV * kH);
  k_init_hidden<<<kH / 8, 64, 0, stream>>>(latent, Wproj, bproj, hA);
  k_init_tok<<<1, 64, 0, stream>>>(tok);

  // Sequential decode: token dependency forces per-step kernels (graph-friendly)
  float* hbufs[2] = {hA, hB};
  for (int t = 0; t < kT; ++t) {
    const float* hin = hbufs[t & 1];
    float*       hou = hbufs[(t + 1) & 1];
    k_gru<<<kH / 8, 64, 0, stream>>>(emb, tok, Wih, bih, Whh, bhh, hin, hou, hbf);
    k_logits_wmma<<<kV / 16, 32, 0, stream>>>(hbf, wbf, bout, logits);
    k_argmax<<<kB, 256, 0, stream>>>(logits, tok, out, t);
  }
  // T=20 even -> final hidden is back in hA
  k_copy_h<<<(kB * kH) / 256, 256, 0, stream>>>(hbufs[kT & 1], out);
}
